// LST_42554535968957
// MI455X (gfx1250) — matrix operand order repaired and checked by reading, not compiled, not run on hardware
//
#include <hip/hip_runtime.h>
#include <hip/hip_bf16.h>

typedef __attribute__((ext_vector_type(16))) _Float16 v16h;
typedef __attribute__((ext_vector_type(8)))  _Float16 v8h;
typedef __attribute__((ext_vector_type(2)))  _Float16 h2;
typedef __attribute__((ext_vector_type(8)))  float    v8f;

#define EMBED  768
#define MTOK   100
#define MPAD   112      // 7 * 16
#define KJ     128      // padded K for the J GEMM
#define NROWS  65536    // 1024 * 64  (x rows)
#define WAVES_PER_BLOCK 2
#define ROWS_PER_WAVE   32   // two 16-row M-tiles per wave (B-fragment reuse x2)

// per-wave LDS: Xh[2][16][768] f16 + J[2][16][128] f16
#define WAVE_LDS_HALVES (2 * 16 * EMBED + 2 * 16 * KJ)

__device__ __forceinline__ v8f zero8() {
  v8f z;
#pragma unroll
  for (int i = 0; i < 8; ++i) z[i] = 0.0f;
  return z;
}

// Load a 16-half WMMA fragment as two 16-byte vector loads (-> *_load_b128).
// lo half at p[0..7], hi half at p[hi_off..hi_off+7].
__device__ __forceinline__ v16h load_frag(const _Float16* p, int hi_off) {
  v8h lo = *(const v8h*)p;
  v8h hi = *(const v8h*)(p + hi_off);
  v16h r;
#pragma unroll
  for (int i = 0; i < 8; ++i) { r[i] = lo[i]; r[8 + i] = hi[i]; }
  return r;
}

// ---------------- prep kernels ----------------

// tokens = tokens_a[layer] @ tokens_b[layer]; f32 + f16 copy padded to 112 rows
__global__ void prep_tokens(const float* __restrict__ ta, const float* __restrict__ tb,
                            const int* __restrict__ layer_p,
                            float* __restrict__ tokens_f32, _Float16* __restrict__ tokens_h) {
  int idx = blockIdx.x * blockDim.x + threadIdx.x;
  if (idx >= MPAD * EMBED) return;
  int m = idx / EMBED, c = idx % EMBED;
  int layer = layer_p[0];
  float s = 0.0f;
  if (m < MTOK) {
    const float* a = ta + (layer * MTOK + m) * 32;
    const float* b = tb + layer * 32 * EMBED + c;
#pragma unroll
    for (int j = 0; j < 32; ++j) s += a[j] * b[j * EMBED];
    tokens_f32[m * EMBED + c] = s;
  }
  tokens_h[m * EMBED + c] = (_Float16)s;
}

// t1p[k][j] = (1<=k<=99) ? tokens[k] . w1[j,:] + b1[j] : 0    (f32, 128x768)
__global__ void prep_t1p(const float* __restrict__ tokens_f32, const float* __restrict__ w1,
                         const float* __restrict__ b1, float* __restrict__ t1p) {
  int idx = blockIdx.x * blockDim.x + threadIdx.x;
  if (idx >= KJ * EMBED) return;
  int k = idx / EMBED, j = idx % EMBED;
  float s = 0.0f;
  if (k >= 1 && k <= 99) {
    s = b1[j];
    const float* tr = tokens_f32 + k * EMBED;
    const float* wr = w1 + j * EMBED;
    for (int q = 0; q < EMBED; ++q) s += tr[q] * wr[q];
  }
  t1p[k * EMBED + j] = s;
}

// WT[c][k] = sum_j t1p[k][j] * w2[c][j]   (f16, 768x128, k contiguous)  == (t1p @ w2^T)^T
__global__ void prep_WT(const float* __restrict__ t1p, const float* __restrict__ w2,
                        _Float16* __restrict__ WT) {
  int idx = blockIdx.x * blockDim.x + threadIdx.x;
  if (idx >= EMBED * KJ) return;
  int c = idx / KJ, k = idx % KJ;
  float s = 0.0f;
  const float* tp = t1p + k * EMBED;
  const float* wr = w2 + c * EMBED;
  for (int q = 0; q < EMBED; ++q) s += tp[q] * wr[q];
  WT[c * KJ + k] = (_Float16)s;
}

__global__ void prep_w2h(const float* __restrict__ w2, _Float16* __restrict__ w2h) {
  int idx = blockIdx.x * blockDim.x + threadIdx.x;
  if (idx < EMBED * EMBED) w2h[idx] = (_Float16)w2[idx];
}

// copy cls token rows (first 64 rows of feats) unchanged
__global__ void copy_cls(const float* __restrict__ feats, float* __restrict__ out) {
  int idx = blockIdx.x * blockDim.x + threadIdx.x;
  if (idx < 64 * EMBED) out[idx] = feats[idx];
}

// ------------- fused main kernel: one wave per TWO 16-row tiles -------------
// out = x + (x @ w2^T + softmax(x @ tokens^T / sqrt(C)) @ W + b2) * scale

__global__ __launch_bounds__(WAVES_PER_BLOCK * 32)
void lst_main(const float* __restrict__ feats, const _Float16* __restrict__ tokens_h,
              const _Float16* __restrict__ WT, const _Float16* __restrict__ w2h,
              const float* __restrict__ b2p, const float* __restrict__ scale_p,
              float* __restrict__ out) {
  extern __shared__ char smem_raw[];

  const int wave = threadIdx.x >> 5;
  const int lane = threadIdx.x & 31;
  const int kh   = lane >> 4;   // wave half
  const int col  = lane & 15;   // N-column within tile / A-row (M)
  const long r0  = (long)(blockIdx.x * WAVES_PER_BLOCK + wave) * ROWS_PER_WAVE;

  const float scale = scale_p[0];
  const float qk = 0.03608439182435161f;  // 768^-0.5

  _Float16* Xh = (_Float16*)smem_raw + wave * WAVE_LDS_HALVES;  // [2][16][768]
  _Float16* Jt = Xh + 2 * 16 * EMBED;                           // [2][16][128]
  const float* xbase = feats + (long)(64 + r0) * EMBED;         // 32 x rows

  // ---- Phase 0: stage X (f32 -> f16) into LDS, coalesced ----
  {
    const float4* src = (const float4*)xbase;
#pragma unroll 4
    for (int i = lane; i < (ROWS_PER_WAVE * EMBED) / 4; i += 32) {
      float4 f = src[i];
      h2 p0, p1;
      p0[0] = (_Float16)f.x; p0[1] = (_Float16)f.y;
      p1[0] = (_Float16)f.z; p1[1] = (_Float16)f.w;
      ((h2*)Xh)[2 * i]     = p0;
      ((h2*)Xh)[2 * i + 1] = p1;
    }
  }
  asm volatile("s_wait_dscnt 0" ::: "memory");

  // ---- GEMM1: S = X @ tokens^T  (K=768, 7 N-tiles, 2 M-tiles) ----
  v8f acc1[2][7];
#pragma unroll
  for (int u = 0; u < 2; ++u)
#pragma unroll
    for (int t = 0; t < 7; ++t) acc1[u][t] = zero8();

#pragma unroll 2
  for (int kc = 0; kc < 24; ++kc) {
    v16h A[2];
#pragma unroll
    for (int u = 0; u < 2; ++u)
      A[u] = load_frag(&Xh[(u * 16 + col) * EMBED + kc * 32 + kh * 8], 16);
#pragma unroll
    for (int t = 0; t < 7; ++t) {
      v16h B = load_frag(tokens_h + (t * 16 + col) * EMBED + kc * 32 + kh * 8, 16);
#pragma unroll
      for (int u = 0; u < 2; ++u)
        acc1[u][t] = __builtin_amdgcn_wmma_f32_16x16x32_f16(false, A[u], false, B,
                                                            (short)0, acc1[u][t], false, false);
    }
  }

  // ---- scale, mask (m >= 100), softmax per row; J -> LDS (f16, K padded to 128) ----
#pragma unroll
  for (int u = 0; u < 2; ++u) {
#pragma unroll
    for (int t = 0; t < 7; ++t)
#pragma unroll
      for (int g = 0; g < 8; ++g) acc1[u][t][g] *= qk;
    if (col >= 4) {
#pragma unroll
      for (int g = 0; g < 8; ++g) acc1[u][6][g] = -1e30f;
    }
#pragma unroll
    for (int g = 0; g < 8; ++g) {
      float mx = acc1[u][0][g];
#pragma unroll
      for (int t = 1; t < 7; ++t) mx = fmaxf(mx, acc1[u][t][g]);
#pragma unroll
      for (int m = 1; m < 16; m <<= 1) mx = fmaxf(mx, __shfl_xor(mx, m, 32));
      float e[7], s = 0.0f;
#pragma unroll
      for (int t = 0; t < 7; ++t) { e[t] = __expf(acc1[u][t][g] - mx); s += e[t]; }
#pragma unroll
      for (int m = 1; m < 16; m <<= 1) s += __shfl_xor(s, m, 32);
      float inv = 1.0f / s;
      int row = g + 8 * kh;
#pragma unroll
      for (int t = 0; t < 7; ++t)
        Jt[u * 16 * KJ + row * KJ + t * 16 + col] = (_Float16)(e[t] * inv);
      Jt[u * 16 * KJ + row * KJ + MPAD + col] = (_Float16)0.0f;
    }
  }
  asm volatile("s_wait_dscnt 0" ::: "memory");

  // ---- Phase 2: acc = J @ W (K=128) + X @ w2^T (K=768); epilogue per n-group ----
  // 12 groups of 4 N-tiles; B fragments shared across the 2 M-tiles.
#pragma unroll 1
  for (int ng = 0; ng < 12; ++ng) {
    // prefetch next group's w2h panel (global_prefetch_b8)
    if (ng + 1 < 12) {
      const char* pf = (const char*)(w2h + (size_t)(ng + 1) * 64 * EMBED);
      __builtin_prefetch(pf + lane * 3072, 0, 1);
    }
    v8f acc[2][4];
#pragma unroll
    for (int u = 0; u < 2; ++u)
#pragma unroll
      for (int j = 0; j < 4; ++j) acc[u][j] = zero8();

    // J @ W  (K = 128)
#pragma unroll 1
    for (int kc = 0; kc < 4; ++kc) {
      v16h Aj[2];
#pragma unroll
      for (int u = 0; u < 2; ++u)
        Aj[u] = load_frag(&Jt[u * 16 * KJ + col * KJ + kc * 32 + kh * 8], 16);
#pragma unroll
      for (int j = 0; j < 4; ++j) {
        int nt = ng * 4 + j;
        v16h B = load_frag(WT + (nt * 16 + col) * KJ + kc * 32 + kh * 8, 16);
#pragma unroll
        for (int u = 0; u < 2; ++u)
          acc[u][j] = __builtin_amdgcn_wmma_f32_16x16x32_f16(false, Aj[u], false, B,
                                                             (short)0, acc[u][j], false, false);
      }
    }

    // X @ w2^T  (K = 768)
#pragma unroll 2
    for (int kc = 0; kc < 24; ++kc) {
      v16h Ax[2];
#pragma unroll
      for (int u = 0; u < 2; ++u)
        Ax[u] = load_frag(&Xh[(u * 16 + col) * EMBED + kc * 32 + kh * 8], 16);
#pragma unroll
      for (int j = 0; j < 4; ++j) {
        int nt = ng * 4 + j;
        v16h B = load_frag(w2h + (nt * 16 + col) * EMBED + kc * 32 + kh * 8, 16);
#pragma unroll
        for (int u = 0; u < 2; ++u)
          acc[u][j] = __builtin_amdgcn_wmma_f32_16x16x32_f16(false, Ax[u], false, B,
                                                             (short)0, acc[u][j], false, false);
      }
    }

    // epilogue: out = x + (acc + b2) * scale
#pragma unroll
    for (int u = 0; u < 2; ++u)
#pragma unroll
      for (int j = 0; j < 4; ++j) {
        int c = (ng * 4 + j) * 16 + col;
        float bv = b2p[c];
#pragma unroll
        for (int g = 0; g < 8; ++g) {
          int row = g + 8 * kh;
          long grow = 64 + r0 + u * 16 + row;
          float xv = feats[grow * EMBED + c];
          out[grow * EMBED + c] = xv + (acc[u][j][g] + bv) * scale;
        }
      }
  }
}

// ---------------- launch ----------------

extern "C" void kernel_launch(void* const* d_in, const int* in_sizes, int n_in,
                              void* d_out, int out_size, void* d_ws, size_t ws_size,
                              hipStream_t stream) {
  (void)in_sizes; (void)n_in; (void)out_size; (void)ws_size;
  const float* feats   = (const float*)d_in[0];
  const float* ta      = (const float*)d_in[1];
  const float* tb      = (const float*)d_in[2];
  const float* scale_p = (const float*)d_in[3];
  const float* w1      = (const float*)d_in[4];
  const float* b1      = (const float*)d_in[5];
  const float* w2      = (const float*)d_in[6];
  const float* b2      = (const float*)d_in[7];
  const int*   layer   = (const int*)d_in[8];
  float* out = (float*)d_out;

  char* ws = (char*)d_ws;
  size_t off = 0;
  float*    tokens_f32 = (float*)(ws + off);    off += (size_t)MTOK * EMBED * 4;   // 307200
  _Float16* tokens_h   = (_Float16*)(ws + off); off += (size_t)MPAD * EMBED * 2;   // 172032
  float*    t1p        = (float*)(ws + off);    off += (size_t)KJ * EMBED * 4;     // 393216
  _Float16* w2h        = (_Float16*)(ws + off); off += (size_t)EMBED * EMBED * 2;  // 1179648
  _Float16* WT         = (_Float16*)(ws + off); off += (size_t)EMBED * KJ * 2;     // 196608

  prep_tokens<<<(MPAD * EMBED + 255) / 256, 256, 0, stream>>>(ta, tb, layer, tokens_f32, tokens_h);
  prep_t1p<<<(KJ * EMBED + 255) / 256, 256, 0, stream>>>(tokens_f32, w1, b1, t1p);
  prep_WT<<<(EMBED * KJ + 255) / 256, 256, 0, stream>>>(t1p, w2, WT);
  prep_w2h<<<(EMBED * EMBED + 255) / 256, 256, 0, stream>>>(w2, w2h);
  copy_cls<<<(64 * EMBED + 255) / 256, 256, 0, stream>>>(feats, out);

  const int waves  = NROWS / ROWS_PER_WAVE;           // 2048
  const int blocks = waves / WAVES_PER_BLOCK;         // 1024
  const size_t lds = (size_t)WAVES_PER_BLOCK * WAVE_LDS_HALVES * sizeof(_Float16); // 114688
  lst_main<<<blocks, WAVES_PER_BLOCK * 32, lds, stream>>>(feats, tokens_h, WT, w2h,
                                                          b2, scale_p, out);
}
